// SelfAttention_48223892800270
// MI455X (gfx1250) — matrix operand order repaired and checked by reading, not compiled, not run on hardware
//
#include <hip/hip_runtime.h>

// Problem constants (match reference)
#define B_ 8
#define D_ 128
#define S_ 4096

typedef __attribute__((ext_vector_type(16))) _Float16 v16h;
typedef __attribute__((ext_vector_type(8)))  float    v8f;

union FragH {
    v16h h;
    unsigned u[8];
};

__device__ __forceinline__ v8f wmma16(const FragH& a, const FragH& b, v8f c) {
    // D = A(16x32 f16) * B(32x16 f16) + C(16x16 f32)
    return __builtin_amdgcn_wmma_f32_16x16x32_f16(
        false, a.h, false, b.h, (short)0, c, false, false);
}

// ---- CDNA5 async global->LDS copy (guarded; falls back to regular copy) ----
#if __has_builtin(__builtin_amdgcn_global_load_async_to_lds_b128) && \
    __has_builtin(__builtin_amdgcn_s_wait_asynccnt)
#define USE_ASYNC_LDS 1
#else
#define USE_ASYNC_LDS 0
#endif

typedef int i4v __attribute__((vector_size(16)));
typedef __attribute__((address_space(1))) i4v* glob_i4p;  // 64-bit global ptr
typedef __attribute__((address_space(3))) i4v* lds_i4p;   // 32-bit LDS ptr

__device__ __forceinline__ void copy16_g2l(const _Float16* gsrc, _Float16* ldst) {
#if USE_ASYNC_LDS
    __builtin_amdgcn_global_load_async_to_lds_b128(
        (glob_i4p)(uintptr_t)gsrc,
        (lds_i4p)(unsigned)(uintptr_t)ldst, 0, 0);
#else
    *(uint4*)ldst = *(const uint4*)gsrc;
#endif
}

__device__ __forceinline__ void wait_async_copies() {
#if USE_ASYNC_LDS
    __builtin_amdgcn_s_wait_asynccnt(0);
#endif
}

// ---------------------------------------------------------------------------
// Kernel 0: convert fp32 weights -> f16 (reused by all projection blocks)
// ---------------------------------------------------------------------------
__global__ __launch_bounds__(256) void k_cvt(
    const float* __restrict__ Wq, const float* __restrict__ Wk,
    const float* __restrict__ Wv,
    _Float16* __restrict__ Wqh, _Float16* __restrict__ Wkh,
    _Float16* __restrict__ Wvh)
{
    int i = blockIdx.x * 256 + threadIdx.x;  // 0 .. D*D-1
    Wqh[i] = (_Float16)Wq[i];
    Wkh[i] = (_Float16)Wk[i];
    Wvh[i] = (_Float16)Wv[i];
}

// ---------------------------------------------------------------------------
// Kernel 1: QKV projection.
//   xt[s][d] = x[b][d][s];  Q = (xt*Wq^T + bq)/sqrt(D), K, V.
//   Q,K stored f16 [b][s][e]; V stored f16 transposed [b][e][s].
// ---------------------------------------------------------------------------
__global__ __launch_bounds__(256) void k_qkv(
    const float* __restrict__ x,
    const _Float16* __restrict__ Wqh, const float* __restrict__ bq,
    const _Float16* __restrict__ Wkh, const float* __restrict__ bk,
    const _Float16* __restrict__ Wvh, const float* __restrict__ bv,
    _Float16* __restrict__ Qh, _Float16* __restrict__ Kh,
    _Float16* __restrict__ Vth)
{
    constexpr int XS = 136;               // halves per LDS row (bank spread)
    __shared__ _Float16 Xt[64 * XS];

    const int tid  = threadIdx.x;
    const int lane = tid & 31;
    const int wv   = tid >> 5;
    const int b    = blockIdx.y;
    const int s0   = blockIdx.x * 64;

    // Stage X tile: Xt[s_local][d] = x[b][d][s0+s_local], fp32 -> f16, coalesced in s.
    for (int i = tid; i < 64 * 128; i += 256) {
        int sl = i & 63;
        int d  = i >> 6;
        Xt[sl * XS + d] = (_Float16)x[((size_t)b * D_ + d) * S_ + (s0 + sl)];
    }
    __syncthreads();

    const int mt = wv & 3;
    const int hf = wv >> 2;

    v8f aq[4], ak[4], av[4];
#pragma unroll
    for (int nt = 0; nt < 4; ++nt) { aq[nt] = {}; ak[nt] = {}; av[nt] = {}; }

    const int arow = mt * 16 + (lane & 15);
    const unsigned* rowd = (const unsigned*)&Xt[arow * XS];

#pragma unroll
    for (int kc = 0; kc < 4; ++kc) {
        // A fragment (16x32 f16): lanes 0-15 hold K0-15; lanes 16-31 K16-31
        FragH a;
        const int ko = kc * 16 + ((lane & 16) ? 8 : 0);
#pragma unroll
        for (int v = 0; v < 4; ++v) {
            a.u[v]     = rowd[ko + v];
            a.u[4 + v] = rowd[ko + 4 + v];
        }
#pragma unroll
        for (int nt = 0; nt < 4; ++nt) {
            const int e  = hf * 64 + nt * 16 + (lane & 15);
            const int kw = kc * 16 + ((lane & 16) ? 8 : 0);
            const unsigned* wq  = (const unsigned*)Wqh + e * 64 + kw;
            const unsigned* wk  = (const unsigned*)Wkh + e * 64 + kw;
            const unsigned* wvv = (const unsigned*)Wvh + e * 64 + kw;
            FragH fq, fk, fv;
#pragma unroll
            for (int v = 0; v < 8; ++v) { fq.u[v] = wq[v]; fk.u[v] = wk[v]; fv.u[v] = wvv[v]; }
            aq[nt] = wmma16(a, fq, aq[nt]);
            ak[nt] = wmma16(a, fk, ak[nt]);
            av[nt] = wmma16(a, fv, av[nt]);
        }
    }

    const float scale = 0.08838834764831845f;  // 1/sqrt(128), folded into Q
#pragma unroll
    for (int nt = 0; nt < 4; ++nt) {
        const int e = hf * 64 + nt * 16 + (lane & 15);
        const float cq = bq[e], ck = bk[e], cv = bv[e];
#pragma unroll
        for (int r = 0; r < 8; ++r) {
            const int s = s0 + mt * 16 + r + ((lane & 16) ? 8 : 0);
            const size_t qi = ((size_t)b * S_ + s) * D_ + e;
            Qh[qi] = (_Float16)((aq[nt][r] + cq) * scale);
            Kh[qi] = (_Float16)(ak[nt][r] + ck);
            Vth[((size_t)b * D_ + e) * S_ + s] = (_Float16)(av[nt][r] + cv);
        }
    }
}

// ---------------------------------------------------------------------------
// Kernel 2: flash attention + residual, output transposed back to (B,D,S).
// Block = 256 threads (8 waves); each wave owns 16 query rows (block = 128 q).
// K/V chunks (32 keys) staged in LDS once per block (double-buffered, async),
// shared by all 8 waves. Softmax row-sums computed with a WMMA against an
// all-ones B fragment (replaces half of the cross-lane shuffle reductions).
// ---------------------------------------------------------------------------
__global__ __launch_bounds__(256) void k_attn(
    const float* __restrict__ x,
    const _Float16* __restrict__ Qh, const _Float16* __restrict__ Kh,
    const _Float16* __restrict__ Vth, float* __restrict__ out)
{
    constexpr int KS = 136;  // halves per K LDS row (32 rows)   -> 272B, 16B-aligned
    constexpr int VS = 40;   // halves per V LDS row (128 rows)  -> 80B,  16B-aligned
    constexpr int PS = 36;   // halves per P row (18 dwords)

    __shared__ __align__(16) _Float16 KT[2][32 * KS];
    __shared__ __align__(16) _Float16 VT[2][128 * VS];
    __shared__ __align__(16) _Float16 P[8 * 16 * PS];

    const int tid  = threadIdx.x;
    const int lane = tid & 31;
    const int wv   = tid >> 5;
    const int b    = blockIdx.y;
    const int q0   = blockIdx.x * 128 + wv * 16;

    // --- stage one 32-key K/V chunk into LDS buffer `buf` (4 x b128 per thread)
    auto stage = [&](int kb, int buf) {
#pragma unroll
        for (int i = 0; i < 2; ++i) {
            const int c = tid + i * 256;              // 0..511
            const int krow = c >> 4, kseg = c & 15;   // K: 32 rows x 16 segs
            copy16_g2l(Kh + ((size_t)b * S_ + kb + krow) * D_ + kseg * 8,
                       &KT[buf][krow * KS + kseg * 8]);
            const int vrow = c >> 2, vseg = c & 3;    // V: 128 rows x 4 segs
            copy16_g2l(Vth + ((size_t)b * D_ + vrow) * S_ + kb + vseg * 8,
                       &VT[buf][vrow * VS + vseg * 8]);
        }
    };

    // Q fragments, held in registers for the whole key sweep.
    FragH aqf[4];
    {
        const int row = q0 + (lane & 15);
        const unsigned* qp = (const unsigned*)(Qh + ((size_t)b * S_ + row) * D_);
#pragma unroll
        for (int kc = 0; kc < 4; ++kc) {
            const int ko = kc * 16 + ((lane & 16) ? 8 : 0);
#pragma unroll
            for (int v = 0; v < 4; ++v) {
                aqf[kc].u[v]     = qp[ko + v];
                aqf[kc].u[4 + v] = qp[ko + 4 + v];
            }
        }
    }

    // All-ones B fragment (f16 1.0 packed): rowsum(P) = P x ones.
    FragH ones;
#pragma unroll
    for (int v = 0; v < 8; ++v) ones.u[v] = 0x3C003C00u;

    v8f o[8];
    float m[8], l[8];
#pragma unroll
    for (int nt = 0; nt < 8; ++nt) o[nt] = {};
#pragma unroll
    for (int r = 0; r < 8; ++r) { m[r] = -3.0e38f; l[r] = 0.0f; }

    _Float16* Pw = &P[wv * 16 * PS];
    const unsigned* Pr = (const unsigned*)Pw;
    const float L2E = 1.4426950408889634f;

    stage(0, 0);

    for (int kb = 0; kb < S_; kb += 32) {
        const int buf = (kb >> 5) & 1;

        // my async copies for `buf` are done; barrier also guarantees every
        // wave finished the previous compute pass on buf^1 before we refill it
        wait_async_copies();
        __syncthreads();
        if (kb + 32 < S_) stage(kb + 32, buf ^ 1);

        // --- scores: two 16-key tiles, S = Q * K^T (scale pre-folded into Q)
        v8f st[2];
#pragma unroll
        for (int j = 0; j < 2; ++j) {
            st[j] = {};
            const unsigned* kp =
                (const unsigned*)&KT[buf][(j * 16 + (lane & 15)) * KS];
#pragma unroll
            for (int kc = 0; kc < 4; ++kc) {
                FragH bkf;
                const int ko = kc * 16 + ((lane & 16) ? 8 : 0);
#pragma unroll
                for (int v = 0; v < 8; ++v) bkf.u[v] = kp[ko + v];
                st[j] = wmma16(aqf[kc], bkf, st[j]);
            }
        }

        // --- online softmax per row; write P tile (f16) to LDS (C->A transpose)
        float corr[8];
#pragma unroll
        for (int r = 0; r < 8; ++r) {
            float rm = fmaxf(st[0][r], st[1][r]);
            rm = fmaxf(rm, __shfl_xor(rm, 1));
            rm = fmaxf(rm, __shfl_xor(rm, 2));
            rm = fmaxf(rm, __shfl_xor(rm, 4));
            rm = fmaxf(rm, __shfl_xor(rm, 8));
            const float mn = fmaxf(m[r], rm);
            corr[r] = exp2f((m[r] - mn) * L2E);
            m[r] = mn;
            const float p0 = exp2f((st[0][r] - mn) * L2E);
            const float p1 = exp2f((st[1][r] - mn) * L2E);
#pragma unroll
            for (int nt = 0; nt < 8; ++nt) o[nt][r] *= corr[r];
            const int prow = r + ((lane & 16) ? 8 : 0);
            const int pcol = lane & 15;
            Pw[prow * PS + pcol]      = (_Float16)p0;
            Pw[prow * PS + 16 + pcol] = (_Float16)p1;
        }

        // Cross-lane LDS RAW inside one wave: wait for the ds stores.
        asm volatile("s_wait_dscnt 0x0" ::: "memory");

        // --- P fragment (A 16x32) read back from LDS
        FragH pa;
        {
            const int prow = lane & 15;
            const int ko = prow * 18 + ((lane & 16) ? 8 : 0);
#pragma unroll
            for (int v = 0; v < 4; ++v) {
                pa.u[v]     = Pr[ko + v];
                pa.u[4 + v] = Pr[ko + 4 + v];
            }
        }

        // --- row sums via WMMA against ones: sums[r] replicated in every col
        {
            v8f sums = {};
            sums = wmma16(pa, ones, sums);
#pragma unroll
            for (int r = 0; r < 8; ++r) l[r] = l[r] * corr[r] + sums[r];
        }

        // --- O += P * V   (V tile in LDS, [e][s]: contiguous K per lane)
#pragma unroll
        for (int nt = 0; nt < 8; ++nt) {
            const unsigned* vp =
                (const unsigned*)&VT[buf][(nt * 16 + (lane & 15)) * VS];
            const int ko = (lane & 16) ? 8 : 0;
            FragH bvf;
#pragma unroll
            for (int v = 0; v < 8; ++v) bvf.u[v] = vp[ko + v];
            o[nt] = wmma16(pa, bvf, o[nt]);
        }
    }

    // --- epilogue: normalize, add residual xt, store transposed (B,D,S)
#pragma unroll
    for (int nt = 0; nt < 8; ++nt) {
        const int e = nt * 16 + (lane & 15);
#pragma unroll
        for (int r = 0; r < 8; ++r) {
            const int s = q0 + r + ((lane & 16) ? 8 : 0);
            const size_t oi = ((size_t)b * D_ + e) * S_ + s;
            out[oi] = x[oi] + o[nt][r] / l[r];
        }
    }
}

// ---------------------------------------------------------------------------
extern "C" void kernel_launch(void* const* d_in, const int* in_sizes, int n_in,
                              void* d_out, int out_size, void* d_ws, size_t ws_size,
                              hipStream_t stream) {
    const float* x  = (const float*)d_in[0];
    const float* Wq = (const float*)d_in[1];
    const float* bq = (const float*)d_in[2];
    const float* Wk = (const float*)d_in[3];
    const float* bk = (const float*)d_in[4];
    const float* Wv = (const float*)d_in[5];
    const float* bv = (const float*)d_in[6];
    float* out = (float*)d_out;

    const size_t BSD = (size_t)B_ * S_ * D_;
    _Float16* Qh  = (_Float16*)d_ws;   // [b][s][e]
    _Float16* Kh  = Qh  + BSD;         // [b][s][e]
    _Float16* Vth = Kh  + BSD;         // [b][e][s]  (transposed)
    _Float16* Wqh = Vth + BSD;
    _Float16* Wkh = Wqh + D_ * D_;
    _Float16* Wvh = Wkh + D_ * D_;

    k_cvt<<<dim3((D_ * D_) / 256), 256, 0, stream>>>(Wq, Wk, Wv, Wqh, Wkh, Wvh);
    k_qkv<<<dim3(S_ / 64, B_), 256, 0, stream>>>(x, Wqh, bq, Wkh, bk, Wvh, bv,
                                                 Qh, Kh, Vth);
    k_attn<<<dim3(S_ / 128, B_), 256, 0, stream>>>(x, Qh, Kh, Vth, out);
}
